// MultiHeadAttention_55482387530306
// MI455X (gfx1250) — matrix operand order repaired and checked by reading, not compiled, not run on hardware
//
#include <hip/hip_runtime.h>

// MI455X / gfx1250 fused multi-head attention forward (wave32, WMMA bf16, TDM).
//
// Shapes: B=2, S=2048, E=512, H=8, dh=64.
// Quirk 1: reshape(b,H,s,-1) w/o transpose -> head blocks are CONTIGUOUS
//          (2048 x 64) slabs of the projected (B,S,E) buffers; same for out.
// Quirk 2: scores scaled by 1/sqrt(s_k)=1/sqrt(2048), folded into Q proj.
//
// ~30 GFLOP vs ~50MB HBM traffic @ 23.3 TB/s -> memory floor ~2us; keep
// scores on-chip (flash attention, online softmax), bf16 WMMA
// (v_wmma_f32_16x16x32_bf16) with fp32 accumulation. One f32->bf16 pre-pass
// keeps the GEMM hot loops conversion-free. K tiles are moved by the Tensor
// Data Mover (tensor_load_to_lds) with hardware LDS padding when available.

typedef __bf16 bf16;
typedef __attribute__((ext_vector_type(16))) __bf16 v16bf;
typedef __attribute__((ext_vector_type(8)))  __bf16 v8bf;
typedef __attribute__((ext_vector_type(8)))  float  v8f;
typedef __attribute__((ext_vector_type(4)))  float  v4f;
typedef __attribute__((ext_vector_type(4)))  unsigned int u32x4;
typedef __attribute__((ext_vector_type(4)))  int    i32x4;
typedef __attribute__((ext_vector_type(8)))  int    i32x8;

#define S_LEN 2048
#define EMB   512
#define N_HEADS 8
#define DH    64
#define BATCH 2
#define KT    64     // keys per flash-attention iteration
#define LDSW  72     // padded LDS row stride (bf16 elems): 144B, 16B-aligned

#if defined(__has_builtin)
#  if __has_builtin(__builtin_amdgcn_tensor_load_to_lds) && \
      __has_builtin(__builtin_amdgcn_s_wait_tensorcnt)
#    define USE_TDM 1
#  endif
#endif
#ifndef USE_TDM
#  define USE_TDM 0
#endif

// Compile-time diagnostics (read back from stderr in the compile loop).
#define CDNA5_STR2(x) #x
#define CDNA5_STR(x) CDNA5_STR2(x)
#pragma message("CDNA5-DIAG clang_major=" CDNA5_STR(__clang_major__))
#if USE_TDM
#pragma message("CDNA5-DIAG TDM path: ENABLED (tensor_load_to_lds + s_wait_tensorcnt)")
#else
#pragma message("CDNA5-DIAG TDM path: DISABLED (cooperative-copy fallback)")
#endif

// ---------------------------------------------------------------------------
// One-shot f32 -> bf16 conversion (8 elems/thread, b128 in / b128 out).
// ---------------------------------------------------------------------------
__global__ __launch_bounds__(256)
void cvt_f32_bf16(const float* __restrict__ in, bf16* __restrict__ out, int n8)
{
    const int i = blockIdx.x * 256 + threadIdx.x;
    if (i < n8) {
        const v4f x0 = *(const v4f*)(in + (size_t)i * 8);
        const v4f x1 = *(const v4f*)(in + (size_t)i * 8 + 4);
        v8bf o;
#pragma unroll
        for (int j = 0; j < 4; ++j) {
            o[j]     = (bf16)x0[j];
            o[4 + j] = (bf16)x1[j];
        }
        *(v8bf*)(out + (size_t)i * 8) = o;
    }
}

// ---------------------------------------------------------------------------
// Projection GEMM (bf16 in, bf16 out): Out = ((X @ W^T) + bias) * scale.
// M = B*S = 4096, N = K = 512. Block = 8 waves; wave = 16(M) x 128(N) tile.
// A frag (16x32): lane<16 holds row m, K {0..15}; lane>=16 {16..31}.
// B frag (32x16): B[k,n] = W[n,k] -> lane = n, contiguous K along the W row.
// ---------------------------------------------------------------------------
__global__ __launch_bounds__(256)
void proj_gemm_bf16(const bf16* __restrict__ X, const bf16* __restrict__ W,
                    const float* __restrict__ bias, bf16* __restrict__ Out,
                    float scale)
{
    const int tid  = threadIdx.x;
    const int w    = tid >> 5;
    const int lane = tid & 31;
    const int hl   = lane >> 4;   // half-wave: 0 or 1
    const int l16  = lane & 15;

    const int m0 = blockIdx.x * 128 + w * 16;
    const int n0 = blockIdx.y * 128;

    v8f zero = {};
    v8f acc[8];
#pragma unroll
    for (int t = 0; t < 8; ++t) acc[t] = zero;

    const bf16* xrow = X + (size_t)(m0 + l16) * EMB;
    for (int k0 = 0; k0 < EMB; k0 += 32) {
        union { v16bf v; v8bf h[2]; } ua;
        ua.h[0] = *(const v8bf*)(xrow + k0 + hl * 16);
        ua.h[1] = *(const v8bf*)(xrow + k0 + hl * 16 + 8);
        const v16bf af = ua.v;
#pragma unroll
        for (int t = 0; t < 8; ++t) {
            const bf16* pb = W + (size_t)(n0 + t * 16 + l16) * EMB + k0 + hl * 16;
            union { v16bf v; v8bf h[2]; } ub;
            ub.h[0] = *(const v8bf*)(pb);
            ub.h[1] = *(const v8bf*)(pb + 8);
            acc[t] = __builtin_amdgcn_wmma_f32_16x16x32_bf16(
                false, af, false, ub.v, (short)0, acc[t], false, false);
        }
    }

    // Epilogue: + bias, * scale, store bf16. C layout: VGPR r, lane<16 -> M=r,
    // lane>=16 -> M=r+8; N = n0 + t*16 + l16.
#pragma unroll
    for (int t = 0; t < 8; ++t) {
        const int n = n0 + t * 16 + l16;
        const float bn = bias[n];
#pragma unroll
        for (int r = 0; r < 8; ++r) {
            const int m = m0 + r + 8 * hl;
            Out[(size_t)m * EMB + n] = (bf16)((acc[t][r] + bn) * scale);
        }
    }
}

// ---------------------------------------------------------------------------
// Flash attention over contiguous (2048 x 64) head slabs.
// Grid: (S/128 q-tiles, B*H). Block: 256 threads = 8 waves, 16 q-rows/wave.
// Per iteration (64 keys): K tile via Tensor Data Mover (HW-padded to the
// 144B LDS row stride) or cooperative fallback, transposed V tile manually,
// S=Q*K^T (8 WMMAs), +mask, online softmax (shfl_xor row reductions within
// 16-lane halves), P via per-wave LDS (C-layout -> A-layout), O += P*V.
// ---------------------------------------------------------------------------
__global__ __launch_bounds__(256)
void attn_fwd_bf16(const bf16* __restrict__ Qp, const bf16* __restrict__ Kp,
                   const bf16* __restrict__ Vp, const float* __restrict__ mask,
                   float* __restrict__ Out)
{
    __shared__ __align__(16) bf16 sK [KT * LDSW];        // K tile, row-major
    __shared__ __align__(16) bf16 sVt[DH * LDSW];        // V tile, transposed
    __shared__ __align__(16) bf16 sP [8 * 16 * LDSW];    // per-wave P tiles

    const int tid  = threadIdx.x;
    const int w    = tid >> 5;
    const int lane = tid & 31;
    const int hl   = lane >> 4;
    const int l16  = lane & 15;

    const size_t hb = (size_t)blockIdx.y * (size_t)S_LEN * DH; // head slab base
    const int m0 = blockIdx.x * 128 + w * 16;                  // q-row base

    // Q fragments: dh=64 -> two K=32 A-fragments, loaded once (scale prefolded).
    v16bf aQ[2];
    {
        const bf16* qrow = Qp + hb + (size_t)(m0 + l16) * DH;
#pragma unroll
        for (int ks = 0; ks < 2; ++ks) {
            union { v16bf v; v8bf h[2]; } u;
            u.h[0] = *(const v8bf*)(qrow + ks * 32 + hl * 16);
            u.h[1] = *(const v8bf*)(qrow + ks * 32 + hl * 16 + 8);
            aQ[ks] = u.v;
        }
    }

    v8f zero = {};
    v8f acc[4];
#pragma unroll
    for (int t = 0; t < 4; ++t) acc[t] = zero;
    float rmax[8], rsum[8];
#pragma unroll
    for (int r = 0; r < 8; ++r) { rmax[r] = -3.0e38f; rsum[r] = 0.0f; }

    bf16* sPw = sP + w * 16 * LDSW;

    for (int j0 = 0; j0 < S_LEN; j0 += KT) {
        __syncthreads();   // previous iteration done reading sK/sVt
        {
            const bf16* kb = Kp + hb + (size_t)j0 * DH;
            const bf16* vb = Vp + hb + (size_t)j0 * DH;
#if USE_TDM
            // Tensor Data Mover: one wave DMAs the 64x64 bf16 K tile into sK.
            // D# per ISA 8.3/8.4: count=1, type=2(image), data_size=2B,
            // tile 64x64, tensor_dim0_stride=64, pad 4 DWORDs per 32 DWORDs
            // -> LDS row stride 128B+16B = 144B = LDSW.
            if (w == 0) {
                const unsigned long long ga = (unsigned long long)(uintptr_t)kb;
                u32x4 g0 = {};
                g0[0] = 1u;                                    // count=1 (user D#)
                g0[1] = (unsigned int)(uintptr_t)&sK[0];       // lds_addr (flat lo32)
                g0[2] = (unsigned int)(ga & 0xFFFFFFFFu);      // global_addr lo
                g0[3] = (unsigned int)((ga >> 32) & 0x1FFFFFFu) | (2u << 30); // hi|type=2
                i32x8 g1 = {};
                g1[0] = (int)((1u << 16)        // data_size = 2 bytes
                            | (1u << 20)        // pad_enable
                            | (4u << 22)        // pad_interval: 32 DWORDs (128B)
                            | (3u << 25));      // pad_amount: 4 DWORDs (16B)
                g1[1] = (int)(64u << 16);       // tensor_dim0 = 64  [79:48]
                g1[2] = (int)((unsigned)S_LEN << 16); // tensor_dim1  [111:80]
                g1[3] = (int)(64u << 16);       // tile_dim0 = 64    [127:112]
                g1[4] = (int)64;                // tile_dim1 = 64    [143:128]
                g1[5] = (int)64;                // tensor_dim0_stride = 64 [207:160]
                i32x4 z4 = {};
#if __clang_major__ >= 23
                i32x8 z8 = {};
                __builtin_amdgcn_tensor_load_to_lds(g0, g1, z4, z4, z8, 0);
#else
                __builtin_amdgcn_tensor_load_to_lds(g0, g1, z4, z4, 0);
#endif
            }
#endif
            // 512 chunks of 8 bf16 (16B); 2 chunks per thread.
            for (int c = tid; c < 512; c += 256) {
                const int row = c >> 3, c8 = (c & 7) * 8;
#if !USE_TDM
                *(v8bf*)&sK[row * LDSW + c8] =
                    *(const v8bf*)(kb + row * DH + c8);
#endif
                v8bf tv = *(const v8bf*)(vb + row * DH + c8);
#pragma unroll
                for (int i = 0; i < 8; ++i)
                    sVt[(c8 + i) * LDSW + row] = tv[i];
            }
            if (j0 + KT < S_LEN)  // prefetch next K tile -> global_prefetch_b8
                __builtin_prefetch(kb + (size_t)KT * DH + tid * 16, 0, 1);
        }
#if USE_TDM
        if (w == 0) __builtin_amdgcn_s_wait_tensorcnt(0);
#endif
        __syncthreads();   // tiles ready

        // ---- S = Q @ K^T  (+ additive mask) ----
        v8f sc[4];
#pragma unroll
        for (int t = 0; t < 4; ++t) {
            const bf16* kr = &sK[(t * 16 + l16) * LDSW];
            union { v16bf v; v8bf h[2]; } u;
            u.h[0] = *(const v8bf*)(kr + hl * 16);
            u.h[1] = *(const v8bf*)(kr + hl * 16 + 8);
            v16bf b0 = u.v;
            u.h[0] = *(const v8bf*)(kr + 32 + hl * 16);
            u.h[1] = *(const v8bf*)(kr + 32 + hl * 16 + 8);
            v16bf b1 = u.v;
            v8f c = zero;
            c = __builtin_amdgcn_wmma_f32_16x16x32_bf16(false, aQ[0], false, b0,
                                                        (short)0, c, false, false);
            c = __builtin_amdgcn_wmma_f32_16x16x32_bf16(false, aQ[1], false, b1,
                                                        (short)0, c, false, false);
            sc[t] = c;
        }
#pragma unroll
        for (int t = 0; t < 4; ++t)
#pragma unroll
            for (int r = 0; r < 8; ++r)
                sc[t][r] += mask[(size_t)(m0 + r + 8 * hl) * S_LEN +
                                 j0 + t * 16 + l16];

        // ---- online softmax update (rows live in 16-lane halves) ----
#pragma unroll
        for (int r = 0; r < 8; ++r) {
            float v = fmaxf(fmaxf(sc[0][r], sc[1][r]), fmaxf(sc[2][r], sc[3][r]));
            v = fmaxf(v, __shfl_xor(v, 1, 32));
            v = fmaxf(v, __shfl_xor(v, 2, 32));
            v = fmaxf(v, __shfl_xor(v, 4, 32));
            v = fmaxf(v, __shfl_xor(v, 8, 32));
            const float nm   = fmaxf(rmax[r], v);
            const float corr = __expf(rmax[r] - nm);
            rmax[r] = nm;
            rsum[r] *= corr;
#pragma unroll
            for (int t = 0; t < 4; ++t) acc[t][r] *= corr;
            float rs = 0.0f;
#pragma unroll
            for (int t = 0; t < 4; ++t) {
                sc[t][r] = __expf(sc[t][r] - nm);
                rs += sc[t][r];
            }
            rs += __shfl_xor(rs, 1, 32);
            rs += __shfl_xor(rs, 2, 32);
            rs += __shfl_xor(rs, 4, 32);
            rs += __shfl_xor(rs, 8, 32);
            rsum[r] += rs;
        }

        // ---- P: C-layout -> A-layout via per-wave LDS (in-order DS, same wave) ----
#pragma unroll
        for (int t = 0; t < 4; ++t)
#pragma unroll
            for (int r = 0; r < 8; ++r)
                sPw[(r + 8 * hl) * LDSW + t * 16 + l16] = (bf16)sc[t][r];
        asm volatile("" ::: "memory");
        v16bf aP[2];
        {
            const bf16* pr = sPw + l16 * LDSW;
#pragma unroll
            for (int ks = 0; ks < 2; ++ks) {
                union { v16bf v; v8bf h[2]; } u;
                u.h[0] = *(const v8bf*)(pr + ks * 32 + hl * 16);
                u.h[1] = *(const v8bf*)(pr + ks * 32 + hl * 16 + 8);
                aP[ks] = u.v;
            }
        }
        asm volatile("" ::: "memory");

        // ---- O += P @ V  (B from transposed V tile: contiguous per lane) ----
#pragma unroll
        for (int t = 0; t < 4; ++t) {
            const bf16* vr = &sVt[(t * 16 + l16) * LDSW];
            union { v16bf v; v8bf h[2]; } u;
            u.h[0] = *(const v8bf*)(vr + hl * 16);
            u.h[1] = *(const v8bf*)(vr + hl * 16 + 8);
            v16bf b0 = u.v;
            u.h[0] = *(const v8bf*)(vr + 32 + hl * 16);
            u.h[1] = *(const v8bf*)(vr + 32 + hl * 16 + 8);
            v16bf b1 = u.v;
            acc[t] = __builtin_amdgcn_wmma_f32_16x16x32_bf16(
                false, aP[0], false, b0, (short)0, acc[t], false, false);
            acc[t] = __builtin_amdgcn_wmma_f32_16x16x32_bf16(
                false, aP[1], false, b1, (short)0, acc[t], false, false);
        }
    }

    // ---- normalize + store fp32 output (head slab is contiguous) ----
#pragma unroll
    for (int r = 0; r < 8; ++r) {
        const float inv = 1.0f / rsum[r];
        const int m = m0 + r + 8 * hl;
#pragma unroll
        for (int t = 0; t < 4; ++t)
            Out[hb + (size_t)m * DH + t * 16 + l16] = acc[t][r] * inv;
    }
}

// ---------------------------------------------------------------------------
extern "C" void kernel_launch(void* const* d_in, const int* in_sizes, int n_in,
                              void* d_out, int out_size, void* d_ws, size_t ws_size,
                              hipStream_t stream)
{
    const float* q    = (const float*)d_in[0];
    const float* k    = (const float*)d_in[1];
    const float* v    = (const float*)d_in[2];
    const float* mask = (const float*)d_in[3];
    const float* Wq   = (const float*)d_in[4];
    const float* bq   = (const float*)d_in[5];
    const float* Wk   = (const float*)d_in[6];
    const float* bk   = (const float*)d_in[7];
    const float* Wv   = (const float*)d_in[8];
    const float* bv   = (const float*)d_in[9];

    const size_t xElems = (size_t)BATCH * S_LEN * EMB;  // 2M
    const size_t wElems = (size_t)EMB * EMB;            // 256K

    bf16* Xq  = (bf16*)d_ws;          // bf16 copies of inputs
    bf16* Xk  = Xq  + xElems;
    bf16* Xv  = Xk  + xElems;
    bf16* Wqb = Xv  + xElems;         // bf16 copies of weights
    bf16* Wkb = Wqb + wElems;
    bf16* Wvb = Wkb + wElems;
    bf16* Qp  = Wvb + wElems;         // bf16 projections
    bf16* Kp  = Qp  + xElems;
    bf16* Vp  = Kp  + xElems;

    // f32 -> bf16 pre-pass (keeps GEMM hot loops conversion-free).
    const int x8 = (int)(xElems / 8), w8 = (int)(wElems / 8);
    cvt_f32_bf16<<<x8 / 256, 256, 0, stream>>>(q, Xq, x8);
    cvt_f32_bf16<<<x8 / 256, 256, 0, stream>>>(k, Xk, x8);
    cvt_f32_bf16<<<x8 / 256, 256, 0, stream>>>(v, Xv, x8);
    cvt_f32_bf16<<<w8 / 256, 256, 0, stream>>>(Wq, Wqb, w8);
    cvt_f32_bf16<<<w8 / 256, 256, 0, stream>>>(Wk, Wkb, w8);
    cvt_f32_bf16<<<w8 / 256, 256, 0, stream>>>(Wv, Wvb, w8);

    const float qscale = 0.022097086912079608f;  // 1/sqrt(2048)

    dim3 gp((BATCH * S_LEN) / 128, EMB / 128);   // (32, 4)
    proj_gemm_bf16<<<gp, 256, 0, stream>>>(Xq, Wqb, bq, Qp, qscale);
    proj_gemm_bf16<<<gp, 256, 0, stream>>>(Xk, Wkb, bk, Kp, 1.0f);
    proj_gemm_bf16<<<gp, 256, 0, stream>>>(Xv, Wvb, bv, Vp, 1.0f);

    dim3 ga(S_LEN / 128, BATCH * N_HEADS);       // (16, 16)
    attn_fwd_bf16<<<ga, 256, 0, stream>>>(Qp, Kp, Vp, mask, (float*)d_out);
}
